// BiLSTM_CRF_34376918237384
// MI455X (gfx1250) — matrix operand order repaired and checked by reading, not compiled, not run on hardware
//
#include <hip/hip_runtime.h>
#include <hip/hip_bf16.h>
#include <math.h>

#define S_LEN   8192
#define D_DIM   256
#define H2_DIM  256
#define G_DIM   1024   // 4*H2
#define H_DIM   512
#define T_TAGS  16
#define START_TAG 0
#define END_TAG   1
#define NEG_INF (-10000.0f)

typedef __attribute__((ext_vector_type(16))) __bf16 v16bf;
typedef __attribute__((ext_vector_type(8)))  float  v8f;

#if __has_builtin(__builtin_amdgcn_global_load_async_to_lds_b128) && \
    __has_builtin(__builtin_amdgcn_s_wait_asynccnt)
#define USE_ASYNC_LDS 1
#endif

#if __has_builtin(__builtin_amdgcn_tensor_load_to_lds) && \
    __has_builtin(__builtin_amdgcn_s_wait_tensorcnt)
#define USE_TDM 1
#endif

__device__ __forceinline__ void cvt4(const float4 f, v16bf& v, int base) {
  v[base + 0] = (__bf16)f.x; v[base + 1] = (__bf16)f.y;
  v[base + 2] = (__bf16)f.z; v[base + 3] = (__bf16)f.w;
}

__device__ __forceinline__ void unpack8(const uint4 q, v16bf& v, int base) {
  unsigned w0 = q.x, w1 = q.y, w2 = q.z, w3 = q.w;
  unsigned short t;
  t = (unsigned short)(w0 & 0xffffu); v[base + 0] = __builtin_bit_cast(__bf16, t);
  t = (unsigned short)(w0 >> 16);     v[base + 1] = __builtin_bit_cast(__bf16, t);
  t = (unsigned short)(w1 & 0xffffu); v[base + 2] = __builtin_bit_cast(__bf16, t);
  t = (unsigned short)(w1 >> 16);     v[base + 3] = __builtin_bit_cast(__bf16, t);
  t = (unsigned short)(w2 & 0xffffu); v[base + 4] = __builtin_bit_cast(__bf16, t);
  t = (unsigned short)(w2 >> 16);     v[base + 5] = __builtin_bit_cast(__bf16, t);
  t = (unsigned short)(w3 & 0xffffu); v[base + 6] = __builtin_bit_cast(__bf16, t);
  t = (unsigned short)(w3 >> 16);     v[base + 7] = __builtin_bit_cast(__bf16, t);
}

__device__ __forceinline__ unsigned lds_offset_of(const void* p) {
  return (unsigned)(unsigned long long)
      (__attribute__((address_space(3))) const char*)p;
}

#if defined(USE_ASYNC_LDS)
typedef int v4i_a __attribute__((ext_vector_type(4)));
__device__ __forceinline__ void async_cp16(const void* g, void* l) {
  __builtin_amdgcn_global_load_async_to_lds_b128(
      (__attribute__((address_space(1))) v4i_a*)(void*)g,
      (__attribute__((address_space(3))) v4i_a*)l, 0, 0);
}
#endif

#if defined(USE_TDM)
typedef unsigned int v4u __attribute__((ext_vector_type(4)));
typedef int          v4i __attribute__((ext_vector_type(4)));
typedef int          v8i __attribute__((ext_vector_type(8)));

// 2D f32 tile load via Tensor Data Mover. D# built per ISA 08_async_tensor §8.
__device__ __forceinline__ void tdm_load_tile_f32(
    const void* gsrc, unsigned lds_byte, unsigned dim0, unsigned rows,
    unsigned stride0) {
  unsigned long long ga = (unsigned long long)(size_t)gsrc;
  v4u g0;
  g0.x = 1u;                                                 // count=1, user desc
  g0.y = lds_byte;                                           // lds_addr (bytes)
  g0.z = (unsigned)ga;                                       // global_addr[31:0]
  g0.w = (unsigned)((ga >> 32) & 0x01ffffffu) | 0x80000000u; // addr[56:32]|type=2
  v8i g1;
  g1[0] = (int)(2u << 16);                                   // data_size=4B
  g1[1] = (int)((dim0 & 0xffffu) << 16);                     // tensor_dim0 lo
  g1[2] = (int)((dim0 >> 16) | ((rows & 0xffffu) << 16));    // dim0 hi | dim1 lo
  g1[3] = (int)(((rows >> 16) & 0xffffu) |
                ((dim0 & 0xffffu) << 16));                   // dim1 hi | tile_dim0
  g1[4] = (int)(rows & 0xffffu);                             // tile_dim1, tile_dim2=0
  g1[5] = (int)stride0;                                      // dim0_stride[31:0]
  g1[6] = 0;                                                 // stride hi bits
  g1[7] = 0;
  v4i z4 = {0, 0, 0, 0};
#if __clang_major__ >= 23
  v8i z8 = {0, 0, 0, 0, 0, 0, 0, 0};
  __builtin_amdgcn_tensor_load_to_lds(g0, g1, z4, z4, z8, 0);
#else
  __builtin_amdgcn_tensor_load_to_lds(g0, g1, z4, z4, 0);
#endif
}
#endif

// ---------------------------------------------------------------------------
// Kernel 1: pre[dir][s][g] = gather(emb) @ W_ih^T + b via bf16 WMMA.
// 4 waves/block share one gathered A tile (staged f32 in LDS by per-lane
// async copies); each wave TDM-loads its own 16x256 W_ih tile into LDS.
// ---------------------------------------------------------------------------
__global__ void __launch_bounds__(128)
gemm_pre_kernel(const int* __restrict__ sentence,
                const float* __restrict__ emb,
                const float* __restrict__ w_ih_f, const float* __restrict__ b_f,
                const float* __restrict__ w_ih_b, const float* __restrict__ b_b,
                float* __restrict__ pre_f, float* __restrict__ pre_b) {
  extern __shared__ float smem[];            // A[16][256] + 4 * B[16][256]
  float* Ash = smem;

  const int tid  = threadIdx.x;
  const int wave = tid >> 5;
  const int lane = tid & 31;
  const int s0   = blockIdx.x * 16;
  const int dir  = blockIdx.z;
  const int g0   = (blockIdx.y * 4 + wave) * 16;

  const float* w_ih = dir ? w_ih_b : w_ih_f;
  const float* bias = dir ? b_b    : b_f;
  float*       pre  = dir ? pre_b  : pre_f;

  // ---- stage gathered A (16 embedding rows, f32, 16KB) ----
#if defined(USE_ASYNC_LDS)
#pragma unroll
  for (int j = 0; j < 8; ++j) {
    const int chunk = tid + j * 128;         // 1024 chunks of 16B
    const int row = chunk >> 6, c4 = (chunk & 63) * 4;
    const int sp = s0 + row;
    const int tp = dir ? (S_LEN - 1 - sp) : sp;
    const float* gsrc = emb + (size_t)sentence[tp] * D_DIM + c4;
    async_cp16(gsrc, Ash + row * D_DIM + c4);
  }
  __builtin_amdgcn_s_wait_asynccnt(0);
#else
#pragma unroll
  for (int j = 0; j < 8; ++j) {
    const int chunk = tid + j * 128;
    const int row = chunk >> 6, c4 = (chunk & 63) * 4;
    const int sp = s0 + row;
    const int tp = dir ? (S_LEN - 1 - sp) : sp;
    *(float4*)(Ash + row * D_DIM + c4) =
        *(const float4*)(emb + (size_t)sentence[tp] * D_DIM + c4);
  }
#endif

  const int kh = lane >> 4;                  // half-wave selector
  const int mn = lane & 15;                  // A-row / B-col within tile

  // ---- this wave's B tile: W_ih rows g0..g0+15 (B[k][n] = w_ih[n][k]) ----
#if defined(USE_TDM)
  float* Bsh = smem + 16 * D_DIM + wave * 16 * D_DIM;
  tdm_load_tile_f32(w_ih + (size_t)g0 * D_DIM, lds_offset_of(Bsh),
                    D_DIM, 16, D_DIM);
  __builtin_amdgcn_s_wait_tensorcnt(0);      // own tile only -> own counter
  const float* bsrc = Bsh + mn * D_DIM;
#else
  const float* bsrc = w_ih + (size_t)(g0 + mn) * D_DIM;
#endif

  __syncthreads();                           // A visible to all waves
  const float* asrc = Ash + mn * D_DIM;

  const int abase = kh ? 8 : 0;              // ISA A layout: lanes>=16 -> K+8
  const int bbase = kh ? 8 : 0;              // ISA B layout: lanes>=16 -> K+8

  v8f acc = {};
  for (int kb = 0; kb < D_DIM; kb += 32) {
    v16bf a, b;
    cvt4(*(const float4*)(asrc + kb + abase),          a, 0);
    cvt4(*(const float4*)(asrc + kb + abase + 4),      a, 4);
    cvt4(*(const float4*)(asrc + kb + 16 + abase),     a, 8);
    cvt4(*(const float4*)(asrc + kb + 16 + abase + 4), a, 12);
    cvt4(*(const float4*)(bsrc + kb + bbase),          b, 0);
    cvt4(*(const float4*)(bsrc + kb + bbase + 4),      b, 4);
    cvt4(*(const float4*)(bsrc + kb + 16 + bbase),     b, 8);
    cvt4(*(const float4*)(bsrc + kb + 16 + bbase + 4), b, 12);
    acc = __builtin_amdgcn_wmma_f32_16x16x32_bf16(
        false, a, false, b, (short)0, acc, false, false);
  }

  const float bv = bias[g0 + mn];
  float* out = pre + (size_t)(s0 + kh * 8) * G_DIM + g0 + mn;  // row = r + 8*kh
#pragma unroll
  for (int r = 0; r < 8; ++r) out[(size_t)r * G_DIM] = acc[r] + bv;
}

// ---------------------------------------------------------------------------
// Kernel 2: sequential LSTM recurrence, one persistent 1024-thread WG per
// direction. h + gate activations in LDS; W_hh streams from L2 in fp32.
// ---------------------------------------------------------------------------
__global__ void __launch_bounds__(1024)
lstm_kernel(const float* __restrict__ pre_f, const float* __restrict__ pre_b,
            const float* __restrict__ w_hh_f, const float* __restrict__ w_hh_b,
            const float* __restrict__ h0_f, const float* __restrict__ c0_f,
            const float* __restrict__ h0_b, const float* __restrict__ c0_b,
            __bf16* __restrict__ hs) {
  const int dir = blockIdx.x;
  const int g   = threadIdx.x;                       // gate row 0..1023
  const float* pre  = dir ? pre_b  : pre_f;
  const float* w_hh = dir ? w_hh_b : w_hh_f;
  const float* h0   = dir ? h0_b   : h0_f;
  const float* c0   = dir ? c0_b   : c0_f;

  __shared__ float h_sh[H2_DIM];
  __shared__ float act[G_DIM];

  float c = 0.0f;
  if (g < H2_DIM) { h_sh[g] = h0[g]; c = c0[g]; }

  const float* wrow = w_hh + (size_t)g * H2_DIM;
  const int grp = g >> 8;                            // 0:i 1:f 2:g 3:o
  const int u   = g & (H2_DIM - 1);

  for (int t = 0; t < S_LEN; ++t) {
    __syncthreads();                                 // h_sh valid
    float a = pre[(size_t)t * G_DIM + g];
#pragma unroll 4
    for (int k = 0; k < H2_DIM; k += 8) {
      float4 wa = *(const float4*)(wrow + k);
      float4 wb = *(const float4*)(wrow + k + 4);
      a += wa.x * h_sh[k + 0] + wa.y * h_sh[k + 1] +
           wa.z * h_sh[k + 2] + wa.w * h_sh[k + 3] +
           wb.x * h_sh[k + 4] + wb.y * h_sh[k + 5] +
           wb.z * h_sh[k + 6] + wb.w * h_sh[k + 7];
    }
    act[g] = (grp == 2) ? tanhf(a) : 1.0f / (1.0f + __expf(-a));
    __syncthreads();                                 // act valid
    if (g < H2_DIM) {
      float gi = act[u], gf = act[H2_DIM + u];
      float gg = act[2 * H2_DIM + u], go = act[3 * H2_DIM + u];
      c = gf * c + gi * gg;
      float h = go * tanhf(c);
      h_sh[u] = h;
      const size_t row = dir ? (size_t)(S_LEN - 1 - t) : (size_t)t;
      hs[row * H_DIM + dir * H2_DIM + u] = (__bf16)h;
    }
  }
}

// ---------------------------------------------------------------------------
// Kernel 3: feats = hs(bf16) @ w_out^T + b_out via WMMA (N=16 is one tile).
// ---------------------------------------------------------------------------
__global__ void __launch_bounds__(32)
outproj_kernel(const __bf16* __restrict__ hs,
               const float* __restrict__ w_out, const float* __restrict__ b_out,
               float* __restrict__ feats) {
  const int lane = threadIdx.x;
  const int s0   = blockIdx.x * 16;
  const int kh   = lane >> 4;
  const int mn   = lane & 15;

  const __bf16* arow = hs    + (size_t)(s0 + mn) * H_DIM;
  const float*  brow = w_out + (size_t)mn * H_DIM;   // B[k][n] = w_out[n][k]
  const int abase = kh ? 8 : 0;
  const int bbase = kh ? 8 : 0;

  v8f acc = {};
  for (int kb = 0; kb < H_DIM; kb += 32) {
    v16bf a, b;
    unpack8(*(const uint4*)(arow + kb + abase),      a, 0);
    unpack8(*(const uint4*)(arow + kb + 16 + abase), a, 8);
    cvt4(*(const float4*)(brow + kb + bbase),        b, 0);
    cvt4(*(const float4*)(brow + kb + bbase + 4),    b, 4);
    cvt4(*(const float4*)(brow + kb + 16 + bbase),     b, 8);
    cvt4(*(const float4*)(brow + kb + 16 + bbase + 4), b, 12);
    acc = __builtin_amdgcn_wmma_f32_16x16x32_bf16(
        false, a, false, b, (short)0, acc, false, false);
  }

  const float bv = b_out[mn];
  float* out = feats + (size_t)(s0 + kh * 8) * T_TAGS + mn;
#pragma unroll
  for (int r = 0; r < 8; ++r) out[(size_t)r * T_TAGS] = acc[r] + bv;
}

// ---------------------------------------------------------------------------
// Kernel 4: Viterbi. One wave32, tag-per-lane, score exchanged with shuffles.
// Backpointers (128 KB) live in LDS so the serial backtrack runs at LDS
// latency — only possible with CDNA5's 320 KB WGP LDS.
// ---------------------------------------------------------------------------
__global__ void __launch_bounds__(32)
viterbi_kernel(const float* __restrict__ feats, const float* __restrict__ trans,
               float* __restrict__ out) {
  extern __shared__ unsigned char bp_sh[];           // S_LEN * T_TAGS bytes
  const int lane = threadIdx.x;
  const int n    = lane & 15;

  float trow[T_TAGS];                                // trans[n][p] cached
#pragma unroll
  for (int p = 0; p < T_TAGS; ++p) trow[p] = trans[n * T_TAGS + p];

  float score = (n == START_TAG) ? 0.0f : NEG_INF;

  for (int s = 0; s < S_LEN; ++s) {
    float best = -INFINITY; int bp = 0;
#pragma unroll
    for (int p = 0; p < T_TAGS; ++p) {
      float v = __shfl(score, p, 32) + trow[p];      // strict > keeps lowest idx
      if (v > best) { best = v; bp = p; }
    }
    bp_sh[s * T_TAGS + n] = (unsigned char)bp;
    score = best + feats[(size_t)s * T_TAGS + n];
    __builtin_prefetch(&feats[(size_t)(s + 1) * T_TAGS], 0, 0);
  }

  score += trans[END_TAG * T_TAGS + n];              // + trans[END]
  float val = (lane < 16) ? score : -INFINITY;
  int   idx = n;
#pragma unroll
  for (int off = 8; off > 0; off >>= 1) {
    float ov = __shfl_down(val, off, 32);
    int   oi = __shfl_down(idx, off, 32);
    if (ov > val || (ov == val && oi < idx)) { val = ov; idx = oi; }
  }

  if (lane == 0) {
    out[0] = val;                                    // path_score
    int tag = idx;
    for (int s = S_LEN - 1; s >= 0; --s) {           // path[s-1] = back[s][path[s]]
      out[1 + s] = (float)tag;
      tag = (int)bp_sh[s * T_TAGS + tag];
    }
  }
}

// ---------------------------------------------------------------------------
extern "C" void kernel_launch(void* const* d_in, const int* in_sizes, int n_in,
                              void* d_out, int out_size, void* d_ws, size_t ws_size,
                              hipStream_t stream) {
  const int*   sentence = (const int*)  d_in[0];
  const float* emb      = (const float*)d_in[1];
  const float* w_ih_f   = (const float*)d_in[2];
  const float* w_hh_f   = (const float*)d_in[3];
  const float* b_f      = (const float*)d_in[4];
  const float* w_ih_b   = (const float*)d_in[5];
  const float* w_hh_b   = (const float*)d_in[6];
  const float* b_b      = (const float*)d_in[7];
  const float* h0_f     = (const float*)d_in[8];
  const float* c0_f     = (const float*)d_in[9];
  const float* h0_b     = (const float*)d_in[10];
  const float* c0_b     = (const float*)d_in[11];
  const float* w_out    = (const float*)d_in[12];
  const float* b_out    = (const float*)d_in[13];
  const float* trans    = (const float*)d_in[14];

  char* ws = (char*)d_ws;
  float*  pre_f = (float*)ws;   ws += (size_t)S_LEN * G_DIM * sizeof(float);
  float*  pre_b = (float*)ws;   ws += (size_t)S_LEN * G_DIM * sizeof(float);
  __bf16* hs    = (__bf16*)ws;  ws += (size_t)S_LEN * H_DIM * sizeof(__bf16);
  float*  feats = (float*)ws;   ws += (size_t)S_LEN * T_TAGS * sizeof(float);

  const size_t gemm_lds = (size_t)(16 + 64) * D_DIM * sizeof(float); // 80 KB

  dim3 g1(S_LEN / 16, G_DIM / (16 * 4), 2);
  gemm_pre_kernel<<<g1, 128, gemm_lds, stream>>>(sentence, emb, w_ih_f, b_f,
                                                 w_ih_b, b_b, pre_f, pre_b);
  lstm_kernel<<<2, 1024, 0, stream>>>(pre_f, pre_b, w_hh_f, w_hh_b,
                                      h0_f, c0_f, h0_b, c0_b, hs);
  outproj_kernel<<<S_LEN / 16, 32, 0, stream>>>(hs, w_out, b_out, feats);
  viterbi_kernel<<<1, 32, S_LEN * T_TAGS, stream>>>(feats, trans, (float*)d_out);
}
